// GNN2Edges_50749333570254
// MI455X (gfx1250) — matrix operand order repaired and checked by reading, not compiled, not run on hardware
//
#include <hip/hip_runtime.h>
#include <hip/hip_bf16.h>
#include <cstdint>
#include <cstddef>

// ---------------- problem constants (from reference) ----------------
#define NNODES 10000
#define NEDGES 50000
#define EDIM   16
#define ZDIM   1024          // 2 * 16 * D
#define BIOUT  272
#define HMAX   512
#define NKB    8             // k-outputs per bilinear block (272 = 34 * 8)

typedef __attribute__((ext_vector_type(16))) __bf16 v16bf;
typedef __attribute__((ext_vector_type(8)))  __bf16 v8bf;
typedef __attribute__((ext_vector_type(4)))  __bf16 v4bf;
typedef __attribute__((ext_vector_type(8)))  float  v8f;

union V16U { v16bf v; v8bf h[2]; };

static __device__ __forceinline__ __bf16 f2bf(float f) { return (__bf16)f; }

// ---------------- small utility kernels ----------------
__global__ void zero_f32(float* __restrict__ p, int n) {
  int t = blockIdx.x * blockDim.x + threadIdx.x;
  if (t < n) p[t] = 0.f;
}

// B[K][N] (f32) -> Bt[N][K] (bf16): coalesced stores, strided loads (one-time pass)
__global__ void transpose_bf16(const float* __restrict__ B, __bf16* __restrict__ Bt,
                               int K, int N) {
  int t = blockIdx.x * blockDim.x + threadIdx.x;
  if (t >= K * N) return;
  int n = t / K, k = t - n * K;
  Bt[t] = f2bf(B[(size_t)k * N + n]);
}

// Wbi[272][1024][16] (f32) -> Wbt[272][16][1024] (bf16)
__global__ void transpose_wbi(const float* __restrict__ Wbi, __bf16* __restrict__ Wbt) {
  int t = blockIdx.x * blockDim.x + threadIdx.x;
  if (t >= BIOUT * EDIM * ZDIM) return;
  int i = t & (ZDIM - 1);
  int j = (t >> 10) & (EDIM - 1);
  int k = t >> 14;                      // 16*1024 = 2^14
  Wbt[t] = f2bf(Wbi[((size_t)k * ZDIM + i) * EDIM + j]);
}

// accumulate per-dst sums of edge_attr + counts (self-loop fill_value='mean')
__global__ void mean_accum(const float* __restrict__ ea, const int* __restrict__ dst,
                           float* __restrict__ meanSum, float* __restrict__ cnt) {
  int t = blockIdx.x * blockDim.x + threadIdx.x;
  if (t >= NEDGES * EDIM) return;
  int e = t >> 4, j = t & 15;
  int d = dst[e];
  atomicAdd(&meanSum[d * EDIM + j], ea[t]);
  if (j == 0) atomicAdd(&cnt[d], 1.f);
}

// augmented edge-attr table: rows [0,E) = edge_attr, rows [E,E+N) = mean_attr
__global__ void build_ea_aug(const float* __restrict__ ea, const float* __restrict__ meanSum,
                             const float* __restrict__ cnt, float* __restrict__ eaAug) {
  int t = blockIdx.x * blockDim.x + threadIdx.x;
  if (t >= (NEDGES + NNODES) * EDIM) return;
  int e = t >> 4, j = t & 15;
  if (e < NEDGES) {
    eaAug[t] = ea[t];
  } else {
    int i = e - NEDGES;
    eaAug[t] = meanSum[i * EDIM + j] / fmaxf(cnt[i], 1.f);
  }
}

// ---------------- generic bf16 WMMA GEMM ----------------
// C[M,N] = act( A[M,K](f32) * Bt[N,K](bf16) + bias ), f32 accumulate via
// v_wmma_f32_16x16x32_bf16. Tiles BM=64, BN=64, BK=32; 8 waves, each wave owns
// TWO 16x16 C tiles (same M rows, N and N+32) so the A fragment is reused and
// A-matrix traffic halves vs BN=32. Branchless vectorized staging.
// Requires K % 4 == 0 (holds: 32/256/512/272).
__global__ __launch_bounds__(256) void gemm_bf16(
    const float* __restrict__ A, const __bf16* __restrict__ Bt,
    const float* __restrict__ bias, float* __restrict__ C,
    int M, int N, int K, int relu) {
  __shared__ alignas(16) __bf16 As[64 * 32];   // [m][k]
  __shared__ alignas(16) __bf16 Bs[64 * 32];   // [n][k]
  const int tid  = threadIdx.x;
  const int wid  = tid >> 5, lane = tid & 31;
  const int mi   = (wid & 3) * 16, ni = (wid >> 2) * 16;   // ni in {0,16}; +32 for tile 2
  const int mBase = blockIdx.y * 64, nBase = blockIdx.x * 64;
  const int mrow = lane & 15;
  const int khalf = lane >> 4;

  // staging coordinates (8 contiguous elems per thread for both A and B)
  const int ar = (tid * 8) >> 5, ac0 = (tid * 8) & 31;
  const int agm  = mBase + ar;
  const int agmc = agm < M ? agm : M - 1;
  const int bgn  = nBase + ar;                 // B uses same row mapping (64 rows)
  const int bgnc = bgn < N ? bgn : N - 1;
  const float*  aRow = A  + (size_t)agmc * K;
  const __bf16* bRow = Bt + (size_t)bgnc * K;

  v8f acc0, acc1;
  #pragma unroll
  for (int i = 0; i < 8; ++i) { acc0[i] = 0.f; acc1[i] = 0.f; }

  for (int k0 = 0; k0 < K; k0 += 32) {
    // ---- stage A (8 f32 -> 8 bf16 per thread, branchless) ----
    {
      v8bf av;
      #pragma unroll
      for (int i = 0; i < 2; ++i) {
        int gk  = k0 + ac0 + i * 4;
        int gkc = (gk + 4 <= K) ? gk : K - 4;
        float4 v = *(const float4*)(aRow + gkc);
        bool ok = (agm < M) && (gk + 4 <= K);
        av[i * 4 + 0] = f2bf(ok ? v.x : 0.f);
        av[i * 4 + 1] = f2bf(ok ? v.y : 0.f);
        av[i * 4 + 2] = f2bf(ok ? v.z : 0.f);
        av[i * 4 + 3] = f2bf(ok ? v.w : 0.f);
      }
      *(v8bf*)(&As[ar * 32 + ac0]) = av;
    }
    // ---- stage B (8 bf16 per thread: one 16B load + one ds_store_b128) ----
    {
      int gk  = k0 + ac0;
      int gkc = (gk + 8 <= K) ? gk : K - 8;
      v8bf v = *(const v8bf*)(bRow + gkc);
      bool ok = (bgn < N) && (gk + 8 <= K);    // K%8==0 for all our K
      v8bf bz;
      #pragma unroll
      for (int j = 0; j < 8; ++j) bz[j] = ok ? v[j] : (__bf16)0.f;
      *(v8bf*)(&Bs[ar * 32 + ac0]) = bz;
    }
    __syncthreads();
    // A fragment: lanes 0-15 K 0..15; lanes 16-31 K 16..31
    const __bf16* ap = &As[(mi + mrow) * 32 + khalf * 16];
    V16U ua; ua.h[0] = *(const v8bf*)ap; ua.h[1] = *(const v8bf*)(ap + 8);
    // B fragments: col n = lane&15; lanes 0-15 K 0..15, lanes 16-31 K 16..31
    const __bf16* bp0 = &Bs[(ni + mrow) * 32 + khalf * 16];
    const __bf16* bp1 = &Bs[(ni + 32 + mrow) * 32 + khalf * 16];
    V16U ub0; ub0.h[0] = *(const v8bf*)bp0; ub0.h[1] = *(const v8bf*)(bp0 + 8);
    V16U ub1; ub1.h[0] = *(const v8bf*)bp1; ub1.h[1] = *(const v8bf*)(bp1 + 8);
    acc0 = __builtin_amdgcn_wmma_f32_16x16x32_bf16(false, ua.v, false, ub0.v,
                                                   (short)0, acc0, false, false);
    acc1 = __builtin_amdgcn_wmma_f32_16x16x32_bf16(false, ua.v, false, ub1.v,
                                                   (short)0, acc1, false, false);
    __syncthreads();
  }
  // C layout: VGPR r -> (M=r, lanes 0-15) / (M=8+r, lanes 16-31), N = lane&15
  #pragma unroll
  for (int r = 0; r < 8; ++r) {
    int row = mBase + mi + r + khalf * 8;
    if (row >= M) continue;
    int col0 = nBase + ni + mrow;
    int col1 = col0 + 32;
    if (col0 < N) {
      float v = acc0[r] + (bias ? bias[col0] : 0.f);
      if (relu) v = fmaxf(v, 0.f);
      C[(size_t)row * N + col0] = v;
    }
    if (col1 < N) {
      float v = acc1[r] + (bias ? bias[col1] : 0.f);
      if (relu) v = fmaxf(v, 0.f);
      C[(size_t)row * N + col1] = v;
    }
  }
}

// ---------------- GATv2 edge kernels ----------------
__global__ __launch_bounds__(256) void gat_alpha(
    const float* __restrict__ xl, const float* __restrict__ xr,
    const float* __restrict__ eaAug, const float* __restrict__ We,
    const float* __restrict__ att, const int* __restrict__ srcA,
    const int* __restrict__ dstA, float* __restrict__ alpha,
    unsigned* __restrict__ amaxu, int dout) {
  int w = (blockIdx.x * blockDim.x + threadIdx.x) >> 5;
  int lane = threadIdx.x & 31;
  if (w >= NEDGES + NNODES) return;
  int s, d;
  if (w < NEDGES) { s = srcA[w]; d = dstA[w]; } else { s = d = w - NEDGES; }
  float eav[EDIM];
  #pragma unroll
  for (int j = 0; j < EDIM; ++j) eav[j] = eaAug[w * EDIM + j];
  float acc = 0.f;
  for (int c = lane; c < dout; c += 32) {
    float ee = 0.f;
    #pragma unroll
    for (int j = 0; j < EDIM; ++j) ee += eav[j] * We[j * dout + c];
    float sv = xl[(size_t)s * dout + c] + xr[(size_t)d * dout + c] + ee;
    sv = sv > 0.f ? sv : 0.2f * sv;
    acc += sv * att[c];
  }
  #pragma unroll
  for (int off = 16; off; off >>= 1) acc += __shfl_xor(acc, off, 32);
  if (lane == 0) {
    alpha[w] = acc;
    unsigned u = __float_as_uint(acc);
    u = (u & 0x80000000u) ? ~u : (u | 0x80000000u);   // order-preserving encode
    atomicMax(&amaxu[d], u);
  }
}

__global__ void gat_exp(const float* __restrict__ alpha, const unsigned* __restrict__ amaxu,
                        const int* __restrict__ dstA, float* __restrict__ ex,
                        float* __restrict__ den) {
  int e = blockIdx.x * blockDim.x + threadIdx.x;
  if (e >= NEDGES + NNODES) return;
  int d = (e < NEDGES) ? dstA[e] : e - NEDGES;
  unsigned u = amaxu[d];
  float m = __uint_as_float((u & 0x80000000u) ? (u ^ 0x80000000u) : ~u);
  float v = __expf(alpha[e] - m);
  ex[e] = v;
  atomicAdd(&den[d], v);
}

__global__ __launch_bounds__(256) void gat_agg(
    const float* __restrict__ xl, const float* __restrict__ ex,
    const float* __restrict__ den, const int* __restrict__ srcA,
    const int* __restrict__ dstA, float* __restrict__ hout, int dout) {
  int w = (blockIdx.x * blockDim.x + threadIdx.x) >> 5;
  int lane = threadIdx.x & 31;
  if (w >= NEDGES + NNODES) return;
  int s, d;
  if (w < NEDGES) { s = srcA[w]; d = dstA[w]; } else { s = d = w - NEDGES; }
  float wgt = ex[w] / fmaxf(den[d], 1e-16f);
  for (int c = lane; c < dout; c += 32)
    atomicAdd(&hout[(size_t)d * dout + c], xl[(size_t)s * dout + c] * wgt);
}

__global__ void gat_final(const float* __restrict__ hout, const float* __restrict__ b,
                          float* __restrict__ h, int dout, int relu) {
  int t = blockIdx.x * blockDim.x + threadIdx.x;
  if (t >= NNODES * dout) return;
  float v = hout[t] + b[t % dout];
  if (relu) v = fmaxf(v, 0.f);
  h[t] = v;
}

// ---------------- edge scoring ----------------
// z[e] = concat(h[src], h[dst]) -> bf16
__global__ void build_z(const float* __restrict__ h, const int* __restrict__ src,
                        const int* __restrict__ dst, __bf16* __restrict__ Z) {
  int t = blockIdx.x * blockDim.x + threadIdx.x;
  if (t >= NEDGES * ZDIM) return;
  int e = t >> 10, c = t & (ZDIM - 1);
  int node = (c < HMAX) ? src[e] : dst[e];
  Z[t] = f2bf(h[(size_t)node * HMAX + (c & (HMAX - 1))]);
}

// bi[e,k] = sum_j ea[e,j] * (sum_i z[e,i] * Wbi[k,i,j]) + bbi[k]
// Dominant GEMM (0.45 TFLOP). Key change: NKB=8 k-outputs per wave, so each A
// fragment (z rows, the 102MB streamed operand) is loaded ONCE per 32-K chunk
// and reused by 8 WMMAs -> A traffic drops from 27.8GB to 3.5GB. Inner loop:
// 2 A b128 loads + 16 B b128 loads (WGP$-shared across waves; Wbt = 8.9MB in L2)
// + 8 v_wmma. Barrier-free. 8 f32 accum sets = 64 VGPRs/lane.
__global__ __launch_bounds__(256) void bilinear_wmma(
    const __bf16* __restrict__ Z, const __bf16* __restrict__ Wbt,
    const float* __restrict__ ea, const float* __restrict__ bbi,
    float* __restrict__ bi) {
  const int kBase = blockIdx.x * NKB;      // 0,8,...,264
  const int eBase = blockIdx.y * 128;
  const int tid = threadIdx.x, wid = tid >> 5, lane = tid & 31;
  const int mrow = lane & 15, khalf = lane >> 4;

  int myRow = eBase + wid * 16 + mrow;
  int rowC = myRow < NEDGES ? myRow : NEDGES - 1;   // clamp; masked on store
  const __bf16* zrow = Z + (size_t)rowC * ZDIM + khalf * 16;
  const __bf16* brow = Wbt + ((size_t)kBase * EDIM + mrow) * ZDIM + khalf * 16;

  v8f acc[NKB];
  #pragma unroll
  for (int kk = 0; kk < NKB; ++kk)
    #pragma unroll
    for (int i = 0; i < 8; ++i) acc[kk][i] = 0.f;

  for (int k0 = 0; k0 < ZDIM; k0 += 32) {
    if (k0 + 128 < ZDIM) __builtin_prefetch(zrow + k0 + 128, 0, 0);  // global_prefetch_b8
    V16U ua;
    ua.h[0] = *(const v8bf*)(zrow + k0);
    ua.h[1] = *(const v8bf*)(zrow + k0 + 8);
    #pragma unroll
    for (int kk = 0; kk < NKB; ++kk) {
      const __bf16* bp = brow + (size_t)kk * EDIM * ZDIM + k0;
      V16U ub;
      ub.h[0] = *(const v8bf*)bp;
      ub.h[1] = *(const v8bf*)(bp + 8);
      acc[kk] = __builtin_amdgcn_wmma_f32_16x16x32_bf16(false, ua.v, false, ub.v,
                                                        (short)0, acc[kk], false, false);
    }
  }
  // epilogue: acc[kk][r] = P[m = r + 8*khalf, j = mrow] for k = kBase+kk.
  // ea[edge, j] is k-independent: load once per row, reuse across all 8 k's.
  #pragma unroll
  for (int r = 0; r < 8; ++r) {
    int m = r + khalf * 8;
    int edge = eBase + wid * 16 + m;
    bool ok = edge < NEDGES;
    float eav = ok ? ea[(size_t)edge * EDIM + mrow] : 0.f;
    #pragma unroll
    for (int kk = 0; kk < NKB; ++kk) {
      float t = acc[kk][r] * eav;
      #pragma unroll
      for (int off = 1; off < 16; off <<= 1) t += __shfl_xor(t, off, 16);
      if (mrow == 0 && ok)
        bi[(size_t)edge * BIOUT + kBase + kk] = t + bbi[kBase + kk];
    }
  }
}

// final: sigmoid(m2 @ Wm3 + bm3), K=32
__global__ void mlp_final(const float* __restrict__ m2, const float* __restrict__ Wm3,
                          const float* __restrict__ bm3, float* __restrict__ out) {
  int e = blockIdx.x * blockDim.x + threadIdx.x;
  if (e >= NEDGES) return;
  float s = bm3[0];
  #pragma unroll
  for (int j = 0; j < 32; ++j) s += m2[(size_t)e * 32 + j] * Wm3[j];
  out[e] = 1.f / (1.f + __expf(-s));
}

// ---------------- host launcher ----------------
extern "C" void kernel_launch(void* const* d_in, const int* in_sizes, int n_in,
                              void* d_out, int out_size, void* d_ws, size_t ws_size,
                              hipStream_t stream) {
  (void)in_sizes; (void)n_in; (void)out_size; (void)ws_size;
  const float* x         = (const float*)d_in[0];
  const float* edge_attr = (const float*)d_in[1];
  const int*   ei        = (const int*)d_in[2];
  const int*   src       = ei;             // edge_index[0]
  const int*   dst       = ei + NEDGES;    // edge_index[1]

  const float* Wl[4]  = {(const float*)d_in[3],  (const float*)d_in[8],
                         (const float*)d_in[13], (const float*)d_in[18]};
  const float* Wr[4]  = {(const float*)d_in[4],  (const float*)d_in[9],
                         (const float*)d_in[14], (const float*)d_in[19]};
  const float* We[4]  = {(const float*)d_in[5],  (const float*)d_in[10],
                         (const float*)d_in[15], (const float*)d_in[20]};
  const float* att[4] = {(const float*)d_in[6],  (const float*)d_in[11],
                         (const float*)d_in[16], (const float*)d_in[21]};
  const float* bb[4]  = {(const float*)d_in[7],  (const float*)d_in[12],
                         (const float*)d_in[17], (const float*)d_in[22]};
  const float* Wbi = (const float*)d_in[23];
  const float* bbi = (const float*)d_in[24];
  const float* Wm1 = (const float*)d_in[25];
  const float* bm1 = (const float*)d_in[26];
  const float* Wm2 = (const float*)d_in[27];
  const float* bm2 = (const float*)d_in[28];
  const float* Wm3 = (const float*)d_in[29];
  const float* bm3 = (const float*)d_in[30];
  float* out = (float*)d_out;

  const int din[4]  = {32, 256, 512, 512};
  const int dou[4]  = {256, 512, 512, 512};
  const int EAUG = NEDGES + NNODES;

  // workspace bump allocator (256B aligned)
  char* base = (char*)d_ws;
  size_t off = 0;
  auto take = [&](size_t bytes) -> void* {
    void* p = base + off;
    off += (bytes + 255) & ~(size_t)255;
    return p;
  };
  float*    cnt     = (float*)take((size_t)NNODES * 4);
  float*    meanSum = (float*)take((size_t)NNODES * EDIM * 4);
  float*    eaAug   = (float*)take((size_t)EAUG * EDIM * 4);
  unsigned* amaxu   = (unsigned*)take((size_t)NNODES * 4);
  float*    den     = (float*)take((size_t)NNODES * 4);
  float*    alpha   = (float*)take((size_t)EAUG * 4);
  float*    ex      = (float*)take((size_t)EAUG * 4);
  float*    hacc    = (float*)take((size_t)NNODES * HMAX * 4);
  float*    h       = (float*)take((size_t)NNODES * HMAX * 4);
  float*    xl      = (float*)take((size_t)NNODES * HMAX * 4);
  float*    xr      = (float*)take((size_t)NNODES * HMAX * 4);
  __bf16*   BtA     = (__bf16*)take((size_t)HMAX * HMAX * 2);
  __bf16*   BtB     = (__bf16*)take((size_t)HMAX * HMAX * 2);
  __bf16*   Wbt     = (__bf16*)take((size_t)BIOUT * EDIM * ZDIM * 2);
  __bf16*   Z       = (__bf16*)take((size_t)NEDGES * ZDIM * 2);
  float*    bi      = (float*)take((size_t)NEDGES * BIOUT * 4);
  float*    m2      = (float*)take((size_t)NEDGES * 32 * 4);
  float*    m1      = (float*)Z;   // alias: Z dead once bi computed

  auto blocks = [](long long n, int bs) { return (unsigned)((n + bs - 1) / bs); };

  // ---- self-loop mean edge_attr + augmented ea table ----
  zero_f32<<<blocks(NNODES, 256), 256, 0, stream>>>(cnt, NNODES);
  zero_f32<<<blocks((long long)NNODES * EDIM, 256), 256, 0, stream>>>(meanSum, NNODES * EDIM);
  mean_accum<<<blocks((long long)NEDGES * EDIM, 256), 256, 0, stream>>>(edge_attr, dst, meanSum, cnt);
  build_ea_aug<<<blocks((long long)EAUG * EDIM, 256), 256, 0, stream>>>(edge_attr, meanSum, cnt, eaAug);

  // ---- one-time Wbi transpose (bf16 [k][j][i]) ----
  transpose_wbi<<<blocks((long long)BIOUT * EDIM * ZDIM, 256), 256, 0, stream>>>(Wbi, Wbt);

  // ---- 4 GATv2 layers ----
  const float* hin = x;
  for (int l = 0; l < 4; ++l) {
    int K = din[l], Nc = dou[l];
    transpose_bf16<<<blocks((long long)K * Nc, 256), 256, 0, stream>>>(Wl[l], BtA, K, Nc);
    transpose_bf16<<<blocks((long long)K * Nc, 256), 256, 0, stream>>>(Wr[l], BtB, K, Nc);
    dim3 g((Nc + 63) / 64, (NNODES + 63) / 64);
    gemm_bf16<<<g, 256, 0, stream>>>(hin, BtA, nullptr, xl, NNODES, Nc, K, 0);
    gemm_bf16<<<g, 256, 0, stream>>>(hin, BtB, nullptr, xr, NNODES, Nc, K, 0);
    zero_f32<<<blocks(NNODES, 256), 256, 0, stream>>>((float*)amaxu, NNODES);
    zero_f32<<<blocks(NNODES, 256), 256, 0, stream>>>(den, NNODES);
    zero_f32<<<blocks((long long)NNODES * Nc, 256), 256, 0, stream>>>(hacc, NNODES * Nc);
    gat_alpha<<<blocks((long long)EAUG * 32, 256), 256, 0, stream>>>(
        xl, xr, eaAug, We[l], att[l], src, dst, alpha, amaxu, Nc);
    gat_exp<<<blocks(EAUG, 256), 256, 0, stream>>>(alpha, amaxu, dst, ex, den);
    gat_agg<<<blocks((long long)EAUG * 32, 256), 256, 0, stream>>>(
        xl, ex, den, src, dst, hacc, Nc);
    gat_final<<<blocks((long long)NNODES * Nc, 256), 256, 0, stream>>>(
        hacc, bb[l], h, Nc, l < 3 ? 1 : 0);
    hin = h;
  }

  // ---- edge scoring ----
  build_z<<<blocks((long long)NEDGES * ZDIM, 256), 256, 0, stream>>>(h, src, dst, Z);
  {
    dim3 g(BIOUT / NKB, (NEDGES + 127) / 128);   // 34 x 391
    bilinear_wmma<<<g, 256, 0, stream>>>(Z, Wbt, edge_attr, bbi, bi);
  }
  transpose_bf16<<<blocks((long long)BIOUT * BIOUT, 256), 256, 0, stream>>>(Wm1, BtA, BIOUT, BIOUT);
  {
    dim3 g((BIOUT + 63) / 64, (NEDGES + 63) / 64);
    gemm_bf16<<<g, 256, 0, stream>>>(bi, BtA, bm1, m1, NEDGES, BIOUT, BIOUT, 1);
  }
  transpose_bf16<<<blocks((long long)BIOUT * 32, 256), 256, 0, stream>>>(Wm2, BtB, BIOUT, 32);
  {
    dim3 g(1, (NEDGES + 63) / 64);
    gemm_bf16<<<g, 256, 0, stream>>>(m1, BtB, bm2, m2, NEDGES, 32, BIOUT, 1);
  }
  mlp_final<<<blocks(NEDGES, 256), 256, 0, stream>>>(m2, Wm3, bm3, out);
}
